// Model_22454089023489
// MI455X (gfx1250) — matrix operand order repaired and checked by reading, not compiled, not run on hardware
//
#include <hip/hip_runtime.h>
#include <hip/hip_bf16.h>

typedef __attribute__((ext_vector_type(16))) _Float16 v16h;
typedef __attribute__((ext_vector_type(8)))  _Float16 v8h;
typedef __attribute__((ext_vector_type(8)))  float    v8f;

#define N_TYPES     3
#define N_MAX       12
#define NFEAT       36          // 3 types * 12 radial
#define FPAD        48          // padded to 3 WMMA tiles of 16
#define ATOM_STRIDE 324         // 3*12*9 floats of c per atom
#define CUTOFF_F    3.6f
#define WIDTH_F     0.3f
#define Y00_F       0.28209479177387814f
#define PI_F        3.14159265358979323846f
#define GRAM_CHUNK  256         // atoms per split-K block

// 0.5 / sigma_n^2 with sigma_n = CUTOFF*max(sqrt(n),1)/N_MAX  => sigma^2 = 0.09*max(n,1)
__constant__ float INV2SIG2[N_MAX] = {
    5.5555555556f, 5.5555555556f, 2.7777777778f, 1.8518518519f,
    1.3888888889f, 1.1111111111f, 0.9259259259f, 0.7936507937f,
    0.6944444444f, 0.6172839506f, 0.5555555556f, 0.5050505051f};

// ---------------------------------------------------------------------------
// Stage 1: per-pair scatter  c[i, type_j, n, m] += radial_n(r)*Y_m(u)*fc(r)
// ---------------------------------------------------------------------------
__global__ void soap_pair_kernel(const float* __restrict__ pos,
                                 const int*   __restrict__ types,
                                 const int2*  __restrict__ pairs,
                                 float*       __restrict__ c,
                                 int E)
{
    int e = blockIdx.x * blockDim.x + threadIdx.x;
    if (e >= E) return;
    int2 p = pairs[e];
    int i = p.x, j = p.y;

    float xi = pos[3*i+0], yi = pos[3*i+1], zi = pos[3*i+2];
    float dx = pos[3*j+0] - xi;
    float dy = pos[3*j+1] - yi;
    float dz = pos[3*j+2] - zi;

    float r2 = dx*dx + dy*dy + dz*dz + 1e-12f;
    float r  = sqrtf(r2);
    if (r >= CUTOFF_F) return;            // fc == 0 -> contributes nothing

    const float start = CUTOFF_F - WIDTH_F;
    float fc = 1.0f;
    if (r >= start)
        fc = 0.5f * (1.0f + __cosf(PI_F * (r - start) / WIDTH_F));

    float inv_r = 1.0f / r;
    float ux = dx * inv_r, uy = dy * inv_r, uz = dz * inv_r;

    float Y[9];
    Y[0] = 0.28209479177387814f;
    Y[1] = 0.4886025119029199f * uy;
    Y[2] = 0.4886025119029199f * uz;
    Y[3] = 0.4886025119029199f * ux;
    Y[4] = 1.0925484305920792f * ux * uy;
    Y[5] = 1.0925484305920792f * uy * uz;
    Y[6] = 0.31539156525252005f * (3.0f * uz * uz - 1.0f);
    Y[7] = 1.0925484305920792f * ux * uz;
    Y[8] = 0.5462742152960396f * (ux * ux - uy * uy);

    int tj = types[j];
    float* dst = c + ((size_t)i * N_TYPES + tj) * (N_MAX * 9);

    float rn = 1.0f;
    #pragma unroll
    for (int n = 0; n < N_MAX; ++n) {
        float rad = fc * rn * __expf(-r2 * INV2SIG2[n]);
        #pragma unroll
        for (int m = 0; m < 9; ++m)
            atomicAdd(dst + n * 9 + m, rad * Y[m]);
        rn *= r;
    }
}

// ---------------------------------------------------------------------------
// Stage 2: self term  c[i, type_i, 0, 0] += Y00   (unique slot per atom)
// ---------------------------------------------------------------------------
__global__ void soap_self_kernel(const int* __restrict__ types,
                                 float* __restrict__ c, int N)
{
    int i = blockIdx.x * blockDim.x + threadIdx.x;
    if (i >= N) return;
    c[((size_t)i * N_TYPES + types[i]) * (N_MAX * 9)] += Y00_F;
}

// ---------------------------------------------------------------------------
// Stage 3a: pack c into feature-major f16 matrices, one per angular block l:
//   Xt[f][s] = (f16) c[atom, f, MLO + (s % MCNT)]  with s = atom*MCNT + mi
// rows f in [36,48) and samples s >= N*MCNT are zero (padding) so the WMMA
// consumer needs no bounds checks at all.
// ---------------------------------------------------------------------------
template <int MLO, int MCNT>
__global__ void soap_pack_kernel(const float* __restrict__ c,
                                 _Float16* __restrict__ Xt,
                                 int N, int Spad)
{
    int s = blockIdx.x * blockDim.x + threadIdx.x;
    int f = blockIdx.y;                       // 0..47
    if (s >= Spad) return;
    float v = 0.0f;
    if (f < NFEAT && s < N * MCNT) {
        int atom = s / MCNT;                  // compile-time divisor
        int mi   = s - atom * MCNT;
        v = c[(size_t)atom * ATOM_STRIDE + f * 9 + MLO + mi];
    }
    Xt[(size_t)f * Spad + s] = (_Float16)v;
}

// ---------------------------------------------------------------------------
// Stage 3b: Gram matrices via WMMA.  G_l = Xt_l * Xt_l^T  (feature x feature)
// One wave per (sample-chunk, 16x16 feature-tile). Feature-major layout makes
// every fragment a pair of contiguous 16B loads per lane - branch free.
// Split-K atomic f32 reduction into the padded 48x48 G_l.
// ---------------------------------------------------------------------------
template <int MCNT>
__global__ __launch_bounds__(32)
void soap_gram_kernel(const _Float16* __restrict__ Xt,
                      float* __restrict__ G, int Spad)
{
    const int lane = threadIdx.x;             // full wave, EXEC all ones
    const int comb = blockIdx.y;              // 0..8 : (ta,tb) tile pair
    const int ta   = comb / 3;
    const int tb   = comb % 3;
    const int s0   = blockIdx.x * (GRAM_CHUNK * MCNT);
    const int scnt = min(GRAM_CHUNK * MCNT, Spad - s0);   // multiple of 32

    const int fa = ta * 16 + (lane & 15);     // A-row feature of this lane
    const int fb = tb * 16 + (lane & 15);     // B-col feature of this lane

    // A fragment: lane<16 holds K {0..7,16..23}, lane>=16 holds {8..15,24..31}
    const _Float16* rowA = Xt + (size_t)fa * Spad + s0 + ((lane < 16) ? 0 : 8);
    // B fragment: lane<16 holds K {0..7,16..23}, lane>=16 holds {8..15,24..31}
    const _Float16* rowB = Xt + (size_t)fb * Spad + s0 + ((lane < 16) ? 0 : 8);

    v8f acc = {};
    for (int k = 0; k < scnt; k += 32) {
        __builtin_prefetch(rowB + k + 128, 0, 1);
        union { v16h v; v8h h[2]; } ua;
        ua.h[0] = *(const v8h*)(rowA + k);        // K = base .. base+7
        ua.h[1] = *(const v8h*)(rowA + k + 16);   // K = base+16 .. base+23
        union { v16h v; v8h h[2]; } ub;
        ub.h[0] = *(const v8h*)(rowB + k);
        ub.h[1] = *(const v8h*)(rowB + k + 16);
        v16h Bf = ub.v;
        acc = __builtin_amdgcn_wmma_f32_16x16x32_f16(
                  false, ua.v, false, Bf, (short)0, acc, false, false);
    }

    // D layout: VGPR r -> M = r + (lane<16?0:8), N = lane&15
    const int mrow = (lane < 16) ? 0 : 8;
    const int ncol = lane & 15;
    #pragma unroll
    for (int r = 0; r < 8; ++r) {
        int f1 = ta * 16 + mrow + r;
        int f2 = tb * 16 + ncol;
        atomicAdd(&G[f1 * FPAD + f2], acc[r]);    // padded rows carry zeros
    }
}

// ---------------------------------------------------------------------------
// Stage 4: energy = bias + sum_l (2l+1)^{-1/2} <W_l, G_l>
// weight index layout: w[(f1*36 + f2)*3 + l]
// ---------------------------------------------------------------------------
__global__ void soap_final_kernel(const float* __restrict__ G,
                                  const float* __restrict__ w,
                                  const float* __restrict__ bias,
                                  float* __restrict__ out)
{
    __shared__ float red[256];
    float s = 0.0f;
    for (int idx = threadIdx.x; idx < 3888; idx += 256) {
        int l   = idx % 3;
        int rem = idx / 3;
        int f2  = rem % NFEAT;
        int f1  = rem / NFEAT;
        float inv = (l == 0) ? 1.0f : ((l == 1) ? 0.5773502691896258f
                                               : 0.4472135954999579f);
        s += w[idx] * inv * G[l * (FPAD * FPAD) + f1 * FPAD + f2];
    }
    red[threadIdx.x] = s;
    __syncthreads();
    for (int o = 128; o > 0; o >>= 1) {
        if (threadIdx.x < o) red[threadIdx.x] += red[threadIdx.x + o];
        __syncthreads();
    }
    if (threadIdx.x == 0) out[0] = red[0] + bias[0];
}

// ---------------------------------------------------------------------------
static inline int round_up(int x, int a) { return (x + a - 1) / a * a; }

extern "C" void kernel_launch(void* const* d_in, const int* in_sizes, int n_in,
                              void* d_out, int out_size, void* d_ws, size_t ws_size,
                              hipStream_t stream)
{
    const float* positions  = (const float*)d_in[0];   // (N,3)
    const float* weight     = (const float*)d_in[1];   // (1,3888)
    const float* bias       = (const float*)d_in[2];   // (1,)
    const int*   atom_types = (const int*)  d_in[3];   // (N,)
    const int2*  pairs      = (const int2*) d_in[4];   // (E,2)

    const int N = in_sizes[3];
    const int E = in_sizes[4] / 2;

    // workspace layout: c (f32) | G (f32) | Xt0 | Xt1 | Xt2 (f16, 64B aligned)
    float* c = (float*)d_ws;
    size_t cElems = (size_t)N * ATOM_STRIDE;
    float* G = c + cElems;

    const int S0 = round_up(N * 1, 32);
    const int S1 = round_up(N * 3, 32);
    const int S2 = round_up(N * 5, 32);

    size_t off = ((cElems + 3 * FPAD * FPAD) * sizeof(float) + 63) & ~(size_t)63;
    _Float16* Xt0 = (_Float16*)((char*)d_ws + off);
    _Float16* Xt1 = Xt0 + (size_t)FPAD * S0;
    _Float16* Xt2 = Xt1 + (size_t)FPAD * S1;

    hipMemsetAsync(d_ws, 0, (cElems + 3 * FPAD * FPAD) * sizeof(float), stream);

    soap_pair_kernel<<<(E + 255) / 256, 256, 0, stream>>>(positions, atom_types,
                                                          pairs, c, E);
    soap_self_kernel<<<(N + 255) / 256, 256, 0, stream>>>(atom_types, c, N);

    soap_pack_kernel<0, 1><<<dim3((S0 + 255) / 256, FPAD), 256, 0, stream>>>(c, Xt0, N, S0);
    soap_pack_kernel<1, 3><<<dim3((S1 + 255) / 256, FPAD), 256, 0, stream>>>(c, Xt1, N, S1);
    soap_pack_kernel<4, 5><<<dim3((S2 + 255) / 256, FPAD), 256, 0, stream>>>(c, Xt2, N, S2);

    dim3 g0((S0 + GRAM_CHUNK * 1 - 1) / (GRAM_CHUNK * 1), 9);
    dim3 g1((S1 + GRAM_CHUNK * 3 - 1) / (GRAM_CHUNK * 3), 9);
    dim3 g2((S2 + GRAM_CHUNK * 5 - 1) / (GRAM_CHUNK * 5), 9);
    soap_gram_kernel<1><<<g0, 32, 0, stream>>>(Xt0, G + 0 * FPAD * FPAD, S0);
    soap_gram_kernel<3><<<g1, 32, 0, stream>>>(Xt1, G + 1 * FPAD * FPAD, S1);
    soap_gram_kernel<5><<<g2, 32, 0, stream>>>(Xt2, G + 2 * FPAD * FPAD, S2);

    soap_final_kernel<<<1, 256, 0, stream>>>(G, weight, bias, (float*)d_out);
}
